// Gemma3Attention_26783416057874
// MI455X (gfx1250) — matrix operand order repaired and checked by reading, not compiled, not run on hardware
//
#include <hip/hip_runtime.h>
#include <hip/hip_bf16.h>

// ---------------------------------------------------------------------------
// Gemma3 sliding-window GQA attention for MI455X (gfx1250, wave32, WMMA).
//
// Roofline: ~165 GFLOP of GEMM-like work vs ~150MB of unavoidable HBM traffic
// (6.5us @ 23.3TB/s) -> matrix-op bound. All matmuls run on
// v_wmma_f32_16x16x32_bf16 (fp32 WMMA only exists as 16x16x4 -> 8x less
// K/instr). Activations/weights are packed once into the ISA's native A/B
// fragment lane layouts. GEMMs stage K-step panels into LDS with
// global_load_async_to_lds_b128 (ASYNCcnt) double-buffering; attention keeps
// per-wave clause'd b128 streams.
// ---------------------------------------------------------------------------

typedef __bf16 bf16;
typedef __attribute__((ext_vector_type(16))) __bf16 v16bf;
typedef __attribute__((ext_vector_type(8)))  __bf16 v8bf;
typedef __attribute__((ext_vector_type(8)))  float  v8f;

#define T_SEQ   4096
#define HIDDEN  2560
#define NQH     8
#define NKH     4
#define HD      256
#define WINDOW  1024
#define QSCALE  0.0625f   // 256^-0.5

__device__ __forceinline__ v8f wmma_bf16(v16bf a, v16bf b, v8f c) {
  // D = A(16x32 bf16) * B(32x16 bf16) + C(16x16 f32)
  return __builtin_amdgcn_wmma_f32_16x16x32_bf16(
      /*neg_a=*/false, a, /*neg_b=*/false, b,
      /*c_mod=*/(short)0, c, /*reuse_a=*/false, /*reuse_b=*/false);
}

// Issue one 16-byte async global->LDS transfer (gfx1250 GV mode, ASYNCcnt).
__device__ __forceinline__ void async_g2l_b128(const bf16* g, const bf16* l) {
  unsigned lds_off = (unsigned)(unsigned long long)(uintptr_t)l; // addr[31:0] = LDS offset
  unsigned long long ga = (unsigned long long)(uintptr_t)g;
  asm volatile("global_load_async_to_lds_b128 %0, %1, off"
               :: "v"(lds_off), "v"(ga) : "memory");
}

// ---------------------------------------------------------------------------
// Packing kernels: fp32 row-major -> bf16 WMMA fragment layout.
// A-fragment (16x32, MxK): lane L holds row M=L%16; element e maps to
//   k = (L/16)*8 + (e&7) + (e>=8 ? 16 : 0)
// B-fragment (32x16, KxN): lane L holds col N=L%16; element e maps to
//   k = (L/16)*8 + (e&7) + (e>=8 ? 16 : 0)
// Fragment (mt,kt)/(nt,kt) occupies 512 contiguous bf16 (32 lanes x 16).
// ---------------------------------------------------------------------------

__global__ __launch_bounds__(256) void pack_a_kernel(const float* __restrict__ src,
                                                     bf16* __restrict__ dst,
                                                     int K, long total) {
  long idx = (long)blockIdx.x * 256 + threadIdx.x;
  if (idx >= total) return;
  int  Kt   = K >> 5;
  int  e    = (int)(idx & 15);
  int  lane = (int)((idx >> 4) & 31);
  long frag = idx >> 9;
  int  kt   = (int)(frag % Kt);
  long mt   = frag / Kt;
  int  m    = (int)(mt * 16 + (lane & 15));
  int  k    = kt * 32 + ((lane >> 4) << 3) + (e & 7) + ((e & 8) ? 16 : 0);
  dst[idx] = (bf16)src[(long)m * K + k];
}

__global__ __launch_bounds__(256) void pack_b_kernel(const float* __restrict__ src,
                                                     bf16* __restrict__ dst,
                                                     int K, int N, long total) {
  long idx = (long)blockIdx.x * 256 + threadIdx.x;
  if (idx >= total) return;
  int  Kt   = K >> 5;
  int  e    = (int)(idx & 15);
  int  lane = (int)((idx >> 4) & 31);
  long frag = idx >> 9;
  int  kt   = (int)(frag % Kt);
  long nt   = frag / Kt;
  int  n    = (int)(nt * 16 + (lane & 15));
  int  k    = kt * 32 + ((lane >> 4) << 3) + (e & 7) + ((e & 8) ? 16 : 0);
  dst[idx] = (bf16)src[(long)k * N + n];
}

// ---------------------------------------------------------------------------
// LDS-staged bf16 GEMM: C[M,N](f32,row-major) = Apack * Bpack.
// Block = 8 waves = 128x256 tile; wave tile 64x64 (4x4 fragments).
// Per K-step the block's A panel (8 frags, 8KB) + B panel (16 frags, 16KB)
// are pulled into LDS with async b128 copies, double-buffered.
// ---------------------------------------------------------------------------

// 6 async b128 transfers per thread: 2 for A (8KB), 4 for B (16KB).
__device__ __forceinline__ void async_stage(const bf16* __restrict__ A,
                                            const bf16* __restrict__ B,
                                            int Kt, int mtBlk, int ntBlk,
                                            int kt, int buf, bf16* lds, int tid) {
#pragma unroll
  for (int s = 0; s < 2; ++s) {
    int u = tid + 256 * s;             // 0..511
    int i = u >> 6;                    // A fragment 0..7
    int o = (u & 63) * 8;              // element offset within fragment
    const bf16* g = A + ((long)(mtBlk + i) * Kt + kt) * 512 + o;
    async_g2l_b128(g, &lds[buf * 4096 + i * 512 + o]);
  }
#pragma unroll
  for (int s = 0; s < 4; ++s) {
    int u = tid + 256 * s;             // 0..1023
    int j = u >> 6;                    // B fragment 0..15
    int o = (u & 63) * 8;
    const bf16* g = B + ((long)(ntBlk + j) * Kt + kt) * 512 + o;
    async_g2l_b128(g, &lds[8192 + buf * 8192 + j * 512 + o]);
  }
}

__global__ __launch_bounds__(256) void gemm_bf16_kernel(const bf16* __restrict__ A,
                                                        const bf16* __restrict__ B,
                                                        float* __restrict__ C,
                                                        int N, int Kt) {
  __shared__ bf16 lds[24576];  // A: 2x4096, B: 2x8192 bf16 (48KB)
  int tid  = threadIdx.x;
  int lane = tid & 31;
  int w    = tid >> 5;
  int wm   = w >> 2, wn = w & 3;
  int mtBlk = blockIdx.x * 8;
  int ntBlk = blockIdx.y * 16;
  int mt0  = mtBlk + wm * 4;
  int nt0  = ntBlk + wn * 4;

  v8f acc[4][4];
#pragma unroll
  for (int i = 0; i < 4; ++i)
#pragma unroll
    for (int j = 0; j < 4; ++j)
#pragma unroll
      for (int r = 0; r < 8; ++r) acc[i][j][r] = 0.0f;

  async_stage(A, B, Kt, mtBlk, ntBlk, 0, 0, lds, tid);

  for (int kt = 0; kt < Kt; ++kt) {
    int buf = kt & 1;
    if (kt + 1 < Kt) {
      async_stage(A, B, Kt, mtBlk, ntBlk, kt + 1, buf ^ 1, lds, tid);
      asm volatile("s_wait_asynccnt 0x6" ::: "memory");  // stage kt retired
    } else {
      asm volatile("s_wait_asynccnt 0x0" ::: "memory");
    }
    __syncthreads();  // all waves' stage-kt data visible in LDS

    v16bf a[4], b[4];
#pragma unroll
    for (int i = 0; i < 4; ++i)
      a[i] = *(const v16bf*)&lds[buf * 4096 + (wm * 4 + i) * 512 + lane * 16];
#pragma unroll
    for (int j = 0; j < 4; ++j)
      b[j] = *(const v16bf*)&lds[8192 + buf * 8192 + (wn * 4 + j) * 512 + lane * 16];
#pragma unroll
    for (int i = 0; i < 4; ++i)
#pragma unroll
      for (int j = 0; j < 4; ++j)
        acc[i][j] = wmma_bf16(a[i], b[j], acc[i][j]);

    __syncthreads();  // reads done before stage kt+2 overwrites this buffer
  }

  int half = lane >> 4, c = lane & 15;
#pragma unroll
  for (int i = 0; i < 4; ++i)
#pragma unroll
    for (int j = 0; j < 4; ++j)
#pragma unroll
      for (int r = 0; r < 8; ++r) {
        int row = (mt0 + i) * 16 + r + 8 * half;
        int col = (nt0 + j) * 16 + c;
        C[(long)row * N + col] = acc[i][j][r];
      }
}

// ---------------------------------------------------------------------------
// Per-head RMSNorm + neox RoPE for q and k. One wave per (token, slot);
// slot 0..7 = q head, slot 8..11 = k head. Folds QSCALE into q. Scatters
// directly into WMMA fragment layouts (q -> A layout, k -> B layout).
// ---------------------------------------------------------------------------
__global__ __launch_bounds__(256) void qk_norm_rope_kernel(
    const float* __restrict__ qraw, const float* __restrict__ kraw,
    const int* __restrict__ positions,
    const float* __restrict__ qw, const float* __restrict__ kw,
    bf16* __restrict__ Qp, bf16* __restrict__ Kp) {
  int lane = threadIdx.x & 31;
  int gw   = blockIdx.x * 8 + (threadIdx.x >> 5);
  int t    = gw / 12;
  int slot = gw % 12;
  bool isq = slot < 8;
  const float* src = isq ? (qraw + (long)t * (NQH * HD) + slot * HD)
                         : (kraw + (long)t * (NKH * HD) + (slot - 8) * HD);
  const float* wgt = isq ? qw : kw;

  int d0 = lane * 8;
  float x[8];
  float ss = 0.f;
#pragma unroll
  for (int e = 0; e < 8; ++e) { x[e] = src[d0 + e]; ss += x[e] * x[e]; }
  ss += __shfl_xor(ss, 16, 32);
  ss += __shfl_xor(ss, 8, 32);
  ss += __shfl_xor(ss, 4, 32);
  ss += __shfl_xor(ss, 2, 32);
  ss += __shfl_xor(ss, 1, 32);
  float rinv = rsqrtf(ss * (1.f / 256.f) + 1e-6f);

  float nx[8];
#pragma unroll
  for (int e = 0; e < 8; ++e) nx[e] = x[e] * rinv * (1.f + wgt[d0 + e]);

  float pos = (float)positions[t];
  const float LN_THETA_OVER = 9.210340371976184f / 128.f;  // ln(10000)/128
#pragma unroll
  for (int e = 0; e < 8; ++e) {
    int d = d0 + e;
    int i = d & 127;
    float f  = pos * __expf(-(float)i * LN_THETA_OVER);
    float cs = __cosf(f), sn = __sinf(f);
    float pn = __shfl_xor(nx[e], 16, 32);  // partner element d ^ 128
    float o  = (d < 128) ? (nx[e] * cs - pn * sn) : (nx[e] * cs + pn * sn);
    if (isq) {
      o *= QSCALE;
      int mt = t >> 4, lane_lo = t & 15;
      int kt = d >> 5, w2 = d & 31;
      int hl = (w2 >> 3) & 1;
      int e2 = (w2 & 7) + ((w2 & 16) ? 8 : 0);
      Qp[(((long)(slot * 256 + mt)) * 8 + kt) * 512 + (hl * 16 + lane_lo) * 16 + e2] = (bf16)o;
    } else {
      int kh = slot - 8;
      int nt = t >> 4, lane_lo = t & 15;
      int kt = d >> 5;
      int hl = (d >> 3) & 1;
      int e2 = (d & 7) + ((d & 16) ? 8 : 0);
      Kp[(((long)(kh * 256 + nt)) * 8 + kt) * 512 + (hl * 16 + lane_lo) * 16 + e2] = (bf16)o;
    }
  }
}

// V -> B-fragment layout for the P@V GEMM (K = key index, tiles of 32 keys).
__global__ __launch_bounds__(256) void pack_v_kernel(const float* __restrict__ vraw,
                                                     bf16* __restrict__ Vp) {
  long idx = (long)blockIdx.x * 256 + threadIdx.x;  // T*KH*HD elements
  int t  = (int)(idx >> 10);
  int r  = (int)(idx & 1023);
  int kh = r >> 8;
  int d  = r & 255;
  int kt2 = t >> 5, rem = t & 31;
  int hl = (rem >> 3) & 1, e = (rem & 7) + ((rem & 16) ? 8 : 0);
  int ntd = d >> 4;
  int lane = hl * 16 + (d & 15);
  Vp[(((long)(kh * 128 + kt2)) * 16 + ntd) * 512 + lane * 16 + e] = (bf16)vraw[idx];
}

// ---------------------------------------------------------------------------
// Flash attention: one wave per (head h, 16-row q tile). Keys processed in
// tiles of 32 within the causal/sliding window. S via WMMA (Q A-frags
// resident), online softmax in C-layout registers, P transposed to A-layout
// through per-wave LDS, O += P@V via WMMA. Output scattered into the
// A-fragment layout consumed by the Wo GEMM.
// ---------------------------------------------------------------------------
__global__ __launch_bounds__(256) void attn_kernel(const bf16* __restrict__ Qp,
                                                   const bf16* __restrict__ Kp,
                                                   const bf16* __restrict__ Vp,
                                                   bf16* __restrict__ OA) {
  __shared__ bf16 plds[8 * 512];  // 1KB P-scratch per wave
  int lane = threadIdx.x & 31;
  int wv   = threadIdx.x >> 5;
  int gw   = blockIdx.x * 8 + wv;
  int h    = gw >> 8;      // 8 heads * 256 q-tiles
  int qt   = gw & 255;
  int kh   = h >> 1;       // GQA: 2 q heads per kv head
  bf16* myp = plds + wv * 512;

  v16bf qf[8];
#pragma unroll
  for (int kt = 0; kt < 8; ++kt)
    qf[kt] = *(const v16bf*)(Qp + (((long)(h * 256 + qt)) * 8 + kt) * 512 + lane * 16);

  v8f acc[16];
#pragma unroll
  for (int n = 0; n < 16; ++n)
#pragma unroll
    for (int r = 0; r < 8; ++r) acc[n][r] = 0.f;

  float mrun[8], lrun[8];
#pragma unroll
  for (int r = 0; r < 8; ++r) { mrun[r] = -1e30f; lrun[r] = 0.f; }

  int half = lane >> 4, c = lane & 15;
  int qlo  = qt * 16;
  int jmin = qlo - (WINDOW - 1); if (jmin < 0) jmin = 0;
  int jb0  = jmin & ~31;

  for (int jb = jb0; jb <= qlo + 15; jb += 32) {
    // ---- S = Q K^T for 32 keys (two 16x16 column fragments) ----
    v8f s0, s1;
#pragma unroll
    for (int r = 0; r < 8; ++r) { s0[r] = 0.f; s1[r] = 0.f; }
    const bf16* kb = Kp + (((long)(kh * 256 + (jb >> 4))) * 8) * 512;
#pragma unroll
    for (int kt = 0; kt < 8; ++kt) {
      v16bf b0 = *(const v16bf*)(kb + (long)kt * 512 + lane * 16);
      v16bf b1 = *(const v16bf*)(kb + (long)(kt + 8) * 512 + lane * 16);
      s0 = wmma_bf16(qf[kt], b0, s0);
      s1 = wmma_bf16(qf[kt], b1, s1);
    }

    // ---- mask + online softmax (row reductions via shfl within halves) ----
    float scl[8];
#pragma unroll
    for (int r = 0; r < 8; ++r) {
      int row = qlo + r + 8 * half;
      int c0 = jb + c, c1 = jb + 16 + c;
      float v0 = ((c0 <= row) && (row - c0 < WINDOW)) ? s0[r] : -1e30f;
      float v1 = ((c1 <= row) && (row - c1 < WINDOW)) ? s1[r] : -1e30f;
      float rm = fmaxf(v0, v1);
      rm = fmaxf(rm, __shfl_xor(rm, 1, 32));
      rm = fmaxf(rm, __shfl_xor(rm, 2, 32));
      rm = fmaxf(rm, __shfl_xor(rm, 4, 32));
      rm = fmaxf(rm, __shfl_xor(rm, 8, 32));
      float mn = fmaxf(mrun[r], rm);
      float sc = __expf(mrun[r] - mn);
      float p0 = __expf(v0 - mn);
      float p1 = __expf(v1 - mn);
      float rs = p0 + p1;
      rs += __shfl_xor(rs, 1, 32);
      rs += __shfl_xor(rs, 2, 32);
      rs += __shfl_xor(rs, 4, 32);
      rs += __shfl_xor(rs, 8, 32);
      lrun[r] = lrun[r] * sc + rs;
      mrun[r] = mn;
      scl[r]  = sc;
      // P tile -> LDS (row-major 16x32 bf16) for C->A layout transpose
      myp[(r + 8 * half) * 32 + c]      = (bf16)p0;
      myp[(r + 8 * half) * 32 + 16 + c] = (bf16)p1;
    }
#pragma unroll
    for (int n = 0; n < 16; ++n)
#pragma unroll
      for (int r = 0; r < 8; ++r) acc[n][r] *= scl[r];

    asm volatile("s_wait_dscnt 0x0" ::: "memory");  // cross-lane LDS hazard

    // ---- read P as A-fragment (K = 32 keys) ----
    int ra = lane & 15, hl = lane >> 4;
    v8bf lo = *(const v8bf*)(myp + ra * 32 + hl * 8);
    v8bf hi = *(const v8bf*)(myp + ra * 32 + 16 + hl * 8);
    v16bf pa = __builtin_shufflevector(lo, hi, 0, 1, 2, 3, 4, 5, 6, 7,
                                       8, 9, 10, 11, 12, 13, 14, 15);

    // ---- O += P @ V over all 16 D-column fragments ----
    const bf16* vb = Vp + (((long)(kh * 128 + (jb >> 5))) * 16) * 512;
#pragma unroll
    for (int n = 0; n < 16; ++n) {
      v16bf vfr = *(const v16bf*)(vb + (long)n * 512 + lane * 16);
      acc[n] = wmma_bf16(pa, vfr, acc[n]);
    }
    asm volatile("" ::: "memory");  // keep next tile's LDS stores below loads
  }

  float il[8];
#pragma unroll
  for (int r = 0; r < 8; ++r) il[r] = 1.f / lrun[r];

  // Scatter O into A-fragment layout for the Wo GEMM (K = H*HD = 2048).
#pragma unroll
  for (int n = 0; n < 16; ++n) {
    int d  = n * 16 + c;
    int kk = h * 256 + d;
    int kt = kk >> 5, w2 = kk & 31;
    int hl2 = (w2 >> 3) & 1;
    int e2  = (w2 & 7) + ((w2 & 16) ? 8 : 0);
#pragma unroll
    for (int r = 0; r < 8; ++r) {
      int lane_lo = r + 8 * half;
      OA[(((long)qt * 64 + kt)) * 512 + (hl2 * 16 + lane_lo) * 16 + e2] =
          (bf16)(acc[n][r] * il[r]);
    }
  }
}

// ---------------------------------------------------------------------------
// Workspace layout (bytes; all offsets 1KB-aligned). Total ~170 MB.
// ---------------------------------------------------------------------------
static const size_t OFF_XA   = 0;                       // x A-packed bf16  (20.97MB)
static const size_t OFF_WQB  = OFF_XA   + 20971520UL;   // Wq B-packed      (10.49MB)
static const size_t OFF_WKB  = OFF_WQB  + 10485760UL;   // Wk B-packed      ( 5.24MB)
static const size_t OFF_WVB  = OFF_WKB  + 5242880UL;    // Wv B-packed      ( 5.24MB)
static const size_t OFF_WOB  = OFF_WVB  + 5242880UL;    // Wo B-packed      (10.49MB)
static const size_t OFF_QRAW = OFF_WOB  + 10485760UL;   // q f32 [T,2048]   (33.55MB)
static const size_t OFF_KRAW = OFF_QRAW + 33554432UL;   // k f32 [T,1024]   (16.78MB)
static const size_t OFF_VRAW = OFF_KRAW + 16777216UL;   // v f32 [T,1024]   (16.78MB)
static const size_t OFF_QP   = OFF_VRAW + 16777216UL;   // q bf16 A-packed  (16.78MB)
static const size_t OFF_KP   = OFF_QP   + 16777216UL;   // k bf16 B-packed  ( 8.39MB)
static const size_t OFF_VP   = OFF_KP   + 8388608UL;    // v bf16 B-packed  ( 8.39MB)
static const size_t OFF_OA   = OFF_VP   + 8388608UL;    // attn-out A-packed(16.78MB)

extern "C" void kernel_launch(void* const* d_in, const int* in_sizes, int n_in,
                              void* d_out, int out_size, void* d_ws, size_t ws_size,
                              hipStream_t stream) {
  const float* x   = (const float*)d_in[0];
  const int*   pos = (const int*)  d_in[1];
  const float* Wq  = (const float*)d_in[2];
  const float* Wk  = (const float*)d_in[3];
  const float* Wv  = (const float*)d_in[4];
  const float* Wo  = (const float*)d_in[5];
  const float* qw  = (const float*)d_in[6];
  const float* kw  = (const float*)d_in[7];
  float* out = (float*)d_out;
  char*  ws  = (char*)d_ws;

  bf16*  xA   = (bf16*)(ws + OFF_XA);
  bf16*  WqB  = (bf16*)(ws + OFF_WQB);
  bf16*  WkB  = (bf16*)(ws + OFF_WKB);
  bf16*  WvB  = (bf16*)(ws + OFF_WVB);
  bf16*  WoB  = (bf16*)(ws + OFF_WOB);
  float* qraw = (float*)(ws + OFF_QRAW);
  float* kraw = (float*)(ws + OFF_KRAW);
  float* vraw = (float*)(ws + OFF_VRAW);
  bf16*  Qp   = (bf16*)(ws + OFF_QP);
  bf16*  Kp   = (bf16*)(ws + OFF_KP);
  bf16*  Vp   = (bf16*)(ws + OFF_VP);
  bf16*  OA   = (bf16*)(ws + OFF_OA);

  // 1) pack activations + weights to bf16 fragment layouts
  pack_a_kernel<<<10485760 / 256, 256, 0, stream>>>(x, xA, HIDDEN, 10485760L);
  pack_b_kernel<<< 5242880 / 256, 256, 0, stream>>>(Wq, WqB, HIDDEN, 2048, 5242880L);
  pack_b_kernel<<< 2621440 / 256, 256, 0, stream>>>(Wk, WkB, HIDDEN, 1024, 2621440L);
  pack_b_kernel<<< 2621440 / 256, 256, 0, stream>>>(Wv, WvB, HIDDEN, 1024, 2621440L);
  pack_b_kernel<<< 5242880 / 256, 256, 0, stream>>>(Wo, WoB, 2048, HIDDEN, 5242880L);

  // 2) QKV projections (fp32 accum via bf16 WMMA, async-LDS staged)
  gemm_bf16_kernel<<<dim3(32, 8), 256, 0, stream>>>(xA, WqB, qraw, 2048, 80);
  gemm_bf16_kernel<<<dim3(32, 4), 256, 0, stream>>>(xA, WkB, kraw, 1024, 80);
  gemm_bf16_kernel<<<dim3(32, 4), 256, 0, stream>>>(xA, WvB, vraw, 1024, 80);

  // 3) RMSNorm + RoPE (+score scale folded into q), repack q/k; pack v
  qk_norm_rope_kernel<<<6144, 256, 0, stream>>>(qraw, kraw, pos, qw, kw, Qp, Kp);
  pack_v_kernel<<<16384, 256, 0, stream>>>(vraw, Vp);

  // 4) flash attention with causal + 1024-left sliding window
  attn_kernel<<<256, 256, 0, stream>>>(Qp, Kp, Vp, OA);

  // 5) output projection -> fp32 d_out [4096, 2560]
  gemm_bf16_kernel<<<dim3(32, 10), 256, 0, stream>>>(OA, WoB, out, HIDDEN, 64);
}
